// QuantumWalkRetriever_86543591014913
// MI455X (gfx1250) — compile-verified, matrix operand order checked by reading, not run on hardware
//
#include <hip/hip_runtime.h>
#include <hip/hip_bf16.h>
#include <math.h>

typedef __attribute__((ext_vector_type(16))) _Float16 v16h;
typedef __attribute__((ext_vector_type(8)))  _Float16 v8h;
typedef __attribute__((ext_vector_type(8)))  float    v8f;

#define NN 100000
#define DD 384
#define HH 128
#define KK 5
#define WALK_STEPS 3
#define INIT_AMP 1.4142135e-3f   // 1/sqrt(N*K) = 1/sqrt(500000)

// ---------------------------------------------------------------------------
// Prep 1: W1[:384,:] (f32, row-major 384x128) -> f16 in WMMA B-fragment order.
// Fragment index: ((j*12 + kt)*32 + lane)*16 + h
//   k   = kt*32 + (h/8)*16 + (lane>>4)*8 + (h&7)   (16-bit B layout, wave32)
//   col = j*16 + (lane&15)
// Each GEMM lane then reads its 16-half fragment as one contiguous 32B load.
// ---------------------------------------------------------------------------
__global__ void k_prep_bfrag(const float* __restrict__ W1, _Float16* __restrict__ Bfrag) {
    int idx = blockIdx.x * blockDim.x + threadIdx.x;
    const int total = 8 * 12 * 32 * 16;
    if (idx >= total) return;
    int h    = idx & 15;
    int lane = (idx >> 4) & 31;
    int kt   = (idx >> 9) % 12;
    int j    = (idx >> 9) / 12;
    int k    = kt * 32 + (h >> 3) * 16 + (lane >> 4) * 8 + (h & 7);
    int col  = j * 16 + (lane & 15);
    Bfrag[idx] = (_Float16)W1[k * HH + col];
}

// Prep 2: fold qv @ W1[384:,:] into the bias: bias2[c] = b1[c] + sum_d qv[d]*W1[384+d][c]
__global__ void k_prep_bias2(const float* __restrict__ W1, const float* __restrict__ b1,
                             const float* __restrict__ qv, float* __restrict__ bias2) {
    int c = threadIdx.x;  // 128 threads
    float s = b1[c];
    for (int d = 0; d < DD; ++d) s += qv[d] * W1[(DD + d) * HH + c];
    bias2[c] = s;
}

// ---------------------------------------------------------------------------
// Main MLP: amps = relu(emb @ W1a + bias2) @ W2 + b2
// Block = 256 threads = 8 waves; each wave: 16 rows x 128 cols via WMMA f16.
// LDS: 12 KB per wave; A staging (16x384 f16) then reused for H (16x128 f32).
// ---------------------------------------------------------------------------
__global__ void __launch_bounds__(256) k_mlp(const float* __restrict__ emb,
                                             const _Float16* __restrict__ Bfrag,
                                             const float* __restrict__ bias2,
                                             const float* __restrict__ W2,
                                             const float* __restrict__ b2,
                                             float* __restrict__ amps) {
    __shared__ __align__(16) char smem[8 * 12288];  // 96 KB (of 320 KB/WGP)
    const int lane    = threadIdx.x & 31;
    const int wave    = threadIdx.x >> 5;
    const int laneHi  = lane >> 4;
    const int rl      = lane & 15;
    const int rowBase = blockIdx.x * 128 + wave * 16;

    _Float16* Aw = (_Float16*)(smem + wave * 12288);

    // Stage A tile: 16 rows x 384 cols, f32 -> f16 (coalesced along columns).
    for (int t = lane; t < 16 * DD; t += 32) {
        int r = t / DD, c = t % DD;
        int gr = rowBase + r;
        if (gr >= NN) gr = 0;  // clamp; stores are guarded later
        Aw[t] = (_Float16)emb[gr * DD + c];
    }
    __syncthreads();

    v8f acc[8] = {};  // 8 column tiles x 8 accumulator VGPRs

    // A fragment base for this lane (16-bit A layout: row = lane&15,
    // lanes 16-31 hold K+8 / K+24; two contiguous 8-half runs -> ds_load_b128 x2).
    const _Float16* Abase = Aw + rl * DD + laneHi * 8;

    for (int kt = 0; kt < 12; ++kt) {
        v8h lo = *(const v8h*)(Abase + kt * 32);
        v8h hi = *(const v8h*)(Abase + kt * 32 + 16);
        v16h a;
#pragma unroll
        for (int i = 0; i < 8; ++i) { a[i] = lo[i]; a[8 + i] = hi[i]; }
#pragma unroll
        for (int j = 0; j < 8; ++j) {
            v16h b = *(const v16h*)(Bfrag + (((j * 12 + kt) * 32 + lane) << 4));
            acc[j] = __builtin_amdgcn_wmma_f32_16x16x32_f16(
                false, a, false, b, (short)0, acc[j], false, false);
        }
    }

    __syncthreads();  // A region about to be reused as H
    float* Hw = (float*)(smem + wave * 12288);
    // C/D layout: VGPR v holds row v (lanes 0-15) / row v+8 (lanes 16-31), col = lane&15.
#pragma unroll
    for (int j = 0; j < 8; ++j) {
        int col = j * 16 + rl;
        float bv = bias2[col];
#pragma unroll
        for (int v = 0; v < 8; ++v) {
            int r = v + laneHi * 8;
            float x = acc[j][v] + bv;
            Hw[r * HH + col] = x > 0.f ? x : 0.f;  // relu
        }
    }
    __syncthreads();

    // Tiny second GEMM: amps[row, 0..4] = h . W2[:,k] + b2[k]
    for (int t = lane; t < 16 * KK; t += 32) {
        int r = t / KK, kk = t % KK;
        int gr = rowBase + r;
        if (gr < NN) {
            float s = b2[kk];
            for (int c = 0; c < HH; ++c) s += Hw[r * HH + c] * W2[c * KK + kk];
            amps[gr * KK + kk] = s;
        }
    }
}

// ---------------------------------------------------------------------------
// Quantum walk kernels
// ---------------------------------------------------------------------------
__global__ void k_init_state(float* __restrict__ state) {
    int i = blockIdx.x * blockDim.x + threadIdx.x;
    if (i < NN * KK) state[i] = INIT_AMP;
}

__global__ void k_fill(float* __restrict__ nxt, float* __restrict__ normAcc) {
    int i = blockIdx.x * blockDim.x + threadIdx.x;
    if (i < NN * KK) nxt[i] = 0.f;
    if (i == 0) *normAcc = 0.f;
}

// coin_n @ s = a*(a.s)/||a||^2 (valid) else uniform: all entries sum(s)/K.
// Scatter: next[nbr[n,j], j] += s_p[j]
__global__ void k_step_scatter(const float* __restrict__ amps, const float* __restrict__ state,
                               const int* __restrict__ nbr, float* __restrict__ nxt) {
    int n = blockIdx.x * blockDim.x + threadIdx.x;
    if (n >= NN) return;
    float a[KK], s[KK];
    float an2 = 0.f, ssum = 0.f, dot = 0.f;
#pragma unroll
    for (int j = 0; j < KK; ++j) {
        a[j] = amps[n * KK + j];
        s[j] = state[n * KK + j];
        an2  += a[j] * a[j];
        ssum += s[j];
        dot  += a[j] * s[j];
    }
    float anorm = sqrtf(an2);
    bool valid = (anorm > 0.f) && (anorm < 3.402823466e38f) && (an2 > 0.f);
    float scale = valid ? (dot / an2) : 0.f;
    float uni = ssum * (1.0f / KK);
#pragma unroll
    for (int j = 0; j < KK; ++j) {
        float sp = valid ? a[j] * scale : uni;
        atomicAdd(&nxt[nbr[n * KK + j] * KK + j], sp);
    }
}

__global__ void k_sumsq(const float* __restrict__ x, float* __restrict__ acc) {
    __shared__ float red[256];
    float s = 0.f;
    for (int i = blockIdx.x * blockDim.x + threadIdx.x; i < NN * KK;
         i += gridDim.x * blockDim.x)
        s += x[i] * x[i];
    red[threadIdx.x] = s;
    __syncthreads();
    for (int off = 128; off > 0; off >>= 1) {
        if (threadIdx.x < off) red[threadIdx.x] += red[threadIdx.x + off];
        __syncthreads();
    }
    if (threadIdx.x == 0) atomicAdd(acc, red[0]);
}

__global__ void k_normalize(const float* __restrict__ nxt, const float* __restrict__ acc,
                            float* __restrict__ state) {
    int i = blockIdx.x * blockDim.x + threadIdx.x;
    if (i >= NN * KK) return;
    float norm = sqrtf(*acc);
    state[i] = (norm > 0.f) ? nxt[i] / norm : INIT_AMP;
}

__global__ void k_final(const float* __restrict__ state, float* __restrict__ out) {
    int n = blockIdx.x * blockDim.x + threadIdx.x;
    if (n >= NN) return;
    float s = 0.f;
#pragma unroll
    for (int j = 0; j < KK; ++j) s += fabsf(state[n * KK + j]);
    out[n] = s;
}

// ---------------------------------------------------------------------------
extern "C" void kernel_launch(void* const* d_in, const int* in_sizes, int n_in,
                              void* d_out, int out_size, void* d_ws, size_t ws_size,
                              hipStream_t stream) {
    const float* emb = (const float*)d_in[0];
    const float* qv  = (const float*)d_in[1];
    const float* W1  = (const float*)d_in[2];
    const float* b1  = (const float*)d_in[3];
    const float* W2  = (const float*)d_in[4];
    const float* b2  = (const float*)d_in[5];
    const int*   nbr = (const int*)d_in[6];
    float* out = (float*)d_out;

    // Workspace layout (~6.1 MB total):
    char* ws = (char*)d_ws;
    _Float16* Bfrag = (_Float16*)ws;                 //  98304 B (W1 f16, fragment order)
    float* bias2    = (float*)(ws + 98304);          //    512 B
    float* amps     = (float*)(ws + 98816);          // 2000000 B
    float* state    = (float*)(ws + 2098816);        // 2000000 B
    float* nxt      = (float*)(ws + 4098816);        // 2000000 B
    float* nrm      = (float*)(ws + 6098816);        //      4 B

    k_prep_bfrag<<<(8 * 12 * 32 * 16 + 255) / 256, 256, 0, stream>>>(W1, Bfrag);
    k_prep_bias2<<<1, 128, 0, stream>>>(W1, b1, qv, bias2);
    k_mlp<<<(NN + 127) / 128, 256, 0, stream>>>(emb, Bfrag, bias2, W2, b2, amps);

    k_init_state<<<(NN * KK + 255) / 256, 256, 0, stream>>>(state);
    for (int s = 0; s < WALK_STEPS; ++s) {
        k_fill<<<(NN * KK + 255) / 256, 256, 0, stream>>>(nxt, nrm);
        k_step_scatter<<<(NN + 255) / 256, 256, 0, stream>>>(amps, state, nbr, nxt);
        k_sumsq<<<512, 256, 0, stream>>>(nxt, nrm);
        k_normalize<<<(NN * KK + 255) / 256, 256, 0, stream>>>(nxt, nrm, state);
    }
    k_final<<<(NN + 255) / 256, 256, 0, stream>>>(state, out);
}
